// MambaBlock_41781441855861
// MI455X (gfx1250) — matrix operand order repaired and checked by reading, not compiled, not run on hardware
//
#include <hip/hip_runtime.h>

#define BB 16
#define LL 8192
#define DMODEL 64
#define DINNER 128
#define DSTATE 16
#define NCHUNK 64
#define CHUNK 128
#define MTOT (BB*LL)

typedef __attribute__((ext_vector_type(16))) _Float16 v16h;
typedef __attribute__((ext_vector_type(8)))  _Float16 v8h;
typedef __attribute__((ext_vector_type(8)))  float    v8f;

__device__ __forceinline__ v8f wmma16(v16h a, v16h b, v8f c) {
  // D = A(16x32 f16) * B(32x16 f16) + C(16x16 f32)
  return __builtin_amdgcn_wmma_f32_16x16x32_f16(false, a, false, b, (short)0, c, false, false);
}

// Async global->LDS copy, 16B per lane, tracked by ASYNCcnt (CDNA5).
__device__ __forceinline__ void async_g2l_b128(void* lds, const void* gaddr) {
  unsigned loff = (unsigned)(size_t)lds;  // low 32 bits of generic LDS ptr = LDS byte offset
  asm volatile("global_load_async_to_lds_b128 %0, %1, off"
               :: "v"(loff), "v"(gaddr) : "memory");
}
__device__ __forceinline__ void wait_async0() {
  asm volatile("s_wait_asynccnt 0x0" ::: "memory");
}

// A-fragment from LDS f16 matrix, rows = M (stride halves), K window [kbase, kbase+32)
__device__ __forceinline__ v16h ldsA(const _Float16* base, int stride, int kbase) {
  int l = threadIdx.x & 31;
  const _Float16* row = base + (l & 15) * stride + kbase + ((l >> 4) << 3);
  v8h lo = *(const v8h*)(row);
  v8h hi = *(const v8h*)(row + 16);
  v16h r;
#pragma unroll
  for (int i = 0; i < 8; ++i) { r[i] = lo[i]; r[8 + i] = hi[i]; }
  return r;
}

// B-fragment from N-major (transposed) LDS f16 weights: rows = N, cols = K
__device__ __forceinline__ v16h ldsB(const _Float16* base, int stride, int kbase, int ntile) {
  int l = threadIdx.x & 31;
  const _Float16* row = base + ((l & 15) + ntile * 16) * stride + kbase + ((l >> 4) << 3);
  v8h lo = *(const v8h*)(row);
  v8h hi = *(const v8h*)(row + 16);
  v16h r;
#pragma unroll
  for (int i = 0; i < 8; ++i) { r[i] = lo[i]; r[8 + i] = hi[i]; }
  return r;
}

// A-fragment straight from global f32 row-major, converting to f16
__device__ __forceinline__ v16h gldA(const float* base, int ldg, int kbase) {
  int l = threadIdx.x & 31;
  const float* row = base + (size_t)(l & 15) * ldg + kbase + ((l >> 4) << 3);
  v16h r;
#pragma unroll
  for (int i = 0; i < 8; ++i) { r[i] = (_Float16)row[i]; r[8 + i] = (_Float16)row[16 + i]; }
  return r;
}

// ---------------- Kernel 1: xin = x@W_in + b ; [xc|z] = xin@in_proj ----------------
__global__ void __launch_bounds__(256) k_inproj(
    const float* __restrict__ x, const float* __restrict__ Win,
    const float* __restrict__ bin, const float* __restrict__ Wproj,
    float* __restrict__ xin, float* __restrict__ xc, float* __restrict__ z)
{
  __shared__ __align__(16) _Float16 sWin[64 * 72];
  __shared__ __align__(16) _Float16 sWp[256 * 72];
  __shared__ __align__(16) _Float16 sStage[8 * 16 * 72];
  int tid = threadIdx.x;
  for (int i = tid; i < 64 * 64; i += 256) {
    int k = i >> 6, n = i & 63;
    sWin[n * 72 + k] = (_Float16)Win[k * 64 + n];
  }
  for (int i = tid; i < 64 * 256; i += 256) {
    int k = i >> 8, n = i & 255;
    sWp[n * 72 + k] = (_Float16)Wproj[k * 256 + n];
  }
  __syncthreads();
  int wave = tid >> 5, lane = tid & 31;
  int row0 = blockIdx.x * 128 + wave * 16;
  _Float16* stage = sStage + wave * 16 * 72;
  int n0 = lane & 15, mb = (lane >> 4) * 8;

  v16h a0 = gldA(x + (size_t)row0 * 64, 64, 0);
  v16h a1 = gldA(x + (size_t)row0 * 64, 64, 32);
#pragma unroll
  for (int nt = 0; nt < 4; ++nt) {
    v8f acc = {};
    acc = wmma16(a0, ldsB(sWin, 72, 0, nt), acc);
    acc = wmma16(a1, ldsB(sWin, 72, 32, nt), acc);
    float bi = bin[nt * 16 + n0];
#pragma unroll
    for (int r = 0; r < 8; ++r) {
      float v = acc[r] + bi;
      xin[(size_t)(row0 + mb + r) * 64 + nt * 16 + n0] = v;
      stage[(mb + r) * 72 + nt * 16 + n0] = (_Float16)v;
    }
  }
  v16h s0 = ldsA(stage, 72, 0);
  v16h s1 = ldsA(stage, 72, 32);
  for (int nt = 0; nt < 16; ++nt) {
    v8f acc = {};
    acc = wmma16(s0, ldsB(sWp, 72, 0, nt), acc);
    acc = wmma16(s1, ldsB(sWp, 72, 32, nt), acc);
    float* outp = (nt < 8) ? (xc + (size_t)row0 * 128 + nt * 16)
                           : (z  + (size_t)row0 * 128 + (nt - 8) * 16);
#pragma unroll
    for (int r = 0; r < 8; ++r) outp[(size_t)(mb + r) * 128 + n0] = acc[r];
  }
}

// ------------- Kernel 2: conv+SiLU -> u ; dbc = u@x_proj ; dt = softplus(...) -------------
__global__ void __launch_bounds__(128) k_conv_xproj(
    const float* __restrict__ xc, const float* __restrict__ convw,
    const float* __restrict__ convb, const float* __restrict__ xprojw,
    const float* __restrict__ dtw, const float* __restrict__ dtb,
    float* __restrict__ ug, float* __restrict__ dtg,
    float* __restrict__ Bm, float* __restrict__ Cm)
{
  __shared__ __align__(16) _Float16 sXp[48 * 136];
  __shared__ float sCw[128 * 4];
  __shared__ float sCb[128];
  __shared__ float sDtw[4 * 128];
  __shared__ float sDtb[128];
  __shared__ __align__(16) _Float16 sU[4 * 16 * 136];
  __shared__ float sDbc[4 * 16 * 48];
  int tid = threadIdx.x;
  for (int i = tid; i < 48 * 128; i += 128) {
    int n = i >> 7, k = i & 127;
    sXp[n * 136 + k] = (n < 36) ? (_Float16)xprojw[k * 36 + n] : (_Float16)0.0f;
  }
  for (int i = tid; i < 512; i += 128) { sCw[i] = convw[i]; sDtw[i] = dtw[i]; }
  sCb[tid] = convb[tid];
  sDtb[tid] = dtb[tid];
  __syncthreads();
  int wave = tid >> 5, lane = tid & 31;
  int row0 = blockIdx.x * 64 + wave * 16;
  int b = row0 >> 13, t0 = row0 & (LL - 1);
  _Float16* uS = sU + wave * 16 * 136;
  float* dbc = sDbc + wave * 16 * 48;
  const float* xcb = xc + (size_t)b * LL * 128;
  for (int m = 0; m < 16; ++m) {
    int t = t0 + m;
#pragma unroll
    for (int c = 0; c < 4; ++c) {
      int d = lane + c * 32;
      float acc = sCb[d];
#pragma unroll
      for (int k = 0; k < 4; ++k) {
        int tt = t - 3 + k;
        if (tt >= 0) acc += xcb[(size_t)tt * 128 + d] * sCw[d * 4 + k];
      }
      float uv = acc / (1.0f + __expf(-acc));  // SiLU
      ug[((size_t)b * LL + t) * 128 + d] = uv;
      uS[m * 136 + d] = (_Float16)uv;
    }
  }
  v16h ua0 = ldsA(uS, 136, 0);
  v16h ua1 = ldsA(uS, 136, 32);
  v16h ua2 = ldsA(uS, 136, 64);
  v16h ua3 = ldsA(uS, 136, 96);
  int n0 = lane & 15, mb = (lane >> 4) * 8;
#pragma unroll
  for (int nt = 0; nt < 3; ++nt) {
    v8f acc = {};
    acc = wmma16(ua0, ldsB(sXp, 136, 0, nt), acc);
    acc = wmma16(ua1, ldsB(sXp, 136, 32, nt), acc);
    acc = wmma16(ua2, ldsB(sXp, 136, 64, nt), acc);
    acc = wmma16(ua3, ldsB(sXp, 136, 96, nt), acc);
#pragma unroll
    for (int r = 0; r < 8; ++r) dbc[(mb + r) * 48 + nt * 16 + n0] = acc[r];
  }
  {  // B / C extraction
    int m = lane & 15;
    size_t rbase = ((size_t)b * LL + t0 + m) * 16;
    if (lane < 16) {
#pragma unroll
      for (int s = 0; s < 16; ++s) Bm[rbase + s] = dbc[m * 48 + 4 + s];
    } else {
#pragma unroll
      for (int s = 0; s < 16; ++s) Cm[rbase + s] = dbc[m * 48 + 20 + s];
    }
  }
  {  // dt = softplus(dbc[:, :4] @ dt_proj_w + dt_proj_b)
    int m = lane & 15;
    int dbase = (lane >> 4) * 64;
    float c0 = dbc[m * 48 + 0], c1 = dbc[m * 48 + 1];
    float c2 = dbc[m * 48 + 2], c3 = dbc[m * 48 + 3];
    size_t rowo = ((size_t)b * LL + t0 + m) * 128;
    for (int j = 0; j < 64; ++j) {
      int d = dbase + j;
      float v = c0 * sDtw[d] + c1 * sDtw[128 + d] + c2 * sDtw[256 + d] + c3 * sDtw[384 + d] + sDtb[d];
      dtg[rowo + d] = (v > 20.0f) ? v : log1pf(__expf(v));
    }
  }
}

// ------------- Kernel 3: chunked scan pass 1 (per-chunk decay product + end state) -------------
__global__ void __launch_bounds__(128) k_scan1(
    const float* __restrict__ dtg, const float* __restrict__ ug,
    const float* __restrict__ Bm, const float* __restrict__ Alog,
    float* __restrict__ Pout, float* __restrict__ hendo)
{
  __shared__ __align__(16) float sB[CHUNK * 16];
  int c = blockIdx.x, b = blockIdx.y;
  int d = threadIdx.x;
  size_t base = ((size_t)b * LL + (size_t)c * CHUNK);
  // async global->LDS staging of B chunk (8KB), 16B per lane per issue
  for (int i = d * 4; i < CHUNK * 16; i += 128 * 4)
    async_g2l_b128(sB + i, Bm + base * 16 + i);
  wait_async0();
  __syncthreads();
  float A[16], h[16], P[16];
#pragma unroll
  for (int s = 0; s < 16; ++s) { A[s] = -__expf(Alog[d * 16 + s]); h[s] = 0.f; P[s] = 1.f; }
  const float* dtp = dtg + base * 128 + d;
  const float* up = ug + base * 128 + d;
  for (int t = 0; t < CHUNK; ++t) {
    if (t + 8 < CHUNK) {
      __builtin_prefetch(dtp + (size_t)(t + 8) * 128, 0, 0);
      __builtin_prefetch(up + (size_t)(t + 8) * 128, 0, 0);
    }
    float dtv = dtp[(size_t)t * 128];
    float uv = up[(size_t)t * 128];
    float du = dtv * uv;
#pragma unroll
    for (int s = 0; s < 16; ++s) {
      float dA = __expf(A[s] * dtv);
      h[s] = dA * h[s] + du * sB[t * 16 + s];
      P[s] *= dA;
    }
  }
  size_t ob = (((size_t)b * NCHUNK + c) * 128 + d) * 16;
#pragma unroll
  for (int s = 0; s < 16; ++s) { Pout[ob + s] = P[s]; hendo[ob + s] = h[s]; }
}

// ------------- Kernel 4: combine chunk summaries -> per-chunk incoming states -------------
__global__ void __launch_bounds__(256) k_combine(
    const float* __restrict__ P, const float* __restrict__ hend, float* __restrict__ hin)
{
  int idx = blockIdx.x * 256 + threadIdx.x;  // 0..32767 = b * 2048 + (d*16+s)
  int b = idx >> 11, ds = idx & 2047;
  float h = 0.f;
  for (int c = 0; c < NCHUNK; ++c) {
    size_t a = ((size_t)(b * NCHUNK + c)) * 2048 + ds;
    hin[a] = h;
    h = P[a] * h + hend[a];
  }
}

// ------------- Kernel 5: scan pass 2 with correct h0, fused D-skip + SiLU(z) gate -------------
__global__ void __launch_bounds__(128) k_scan2(
    const float* __restrict__ dtg, const float* __restrict__ ug,
    const float* __restrict__ Bm, const float* __restrict__ Cm,
    const float* __restrict__ Alog, const float* __restrict__ Dskip,
    const float* __restrict__ zg, const float* __restrict__ hinb,
    float* __restrict__ yout)
{
  __shared__ __align__(16) float sB[CHUNK * 16];
  __shared__ __align__(16) float sC[CHUNK * 16];
  int c = blockIdx.x, b = blockIdx.y;
  int d = threadIdx.x;
  size_t base = ((size_t)b * LL + (size_t)c * CHUNK);
  for (int i = d * 4; i < CHUNK * 16; i += 128 * 4) {
    async_g2l_b128(sB + i, Bm + base * 16 + i);
    async_g2l_b128(sC + i, Cm + base * 16 + i);
  }
  wait_async0();
  __syncthreads();
  float A[16], h[16];
  size_t hb = (((size_t)b * NCHUNK + c) * 128 + d) * 16;
#pragma unroll
  for (int s = 0; s < 16; ++s) { A[s] = -__expf(Alog[d * 16 + s]); h[s] = hinb[hb + s]; }
  float Dv = Dskip[d];
  const float* dtp = dtg + base * 128 + d;
  const float* up = ug + base * 128 + d;
  const float* zp = zg + base * 128 + d;
  float* yp = yout + base * 128 + d;
  for (int t = 0; t < CHUNK; ++t) {
    if (t + 8 < CHUNK) {
      __builtin_prefetch(dtp + (size_t)(t + 8) * 128, 0, 0);
      __builtin_prefetch(up + (size_t)(t + 8) * 128, 0, 0);
      __builtin_prefetch(zp + (size_t)(t + 8) * 128, 0, 0);
    }
    float dtv = dtp[(size_t)t * 128];
    float uv = up[(size_t)t * 128];
    float du = dtv * uv;
    float y = 0.f;
#pragma unroll
    for (int s = 0; s < 16; ++s) {
      float dA = __expf(A[s] * dtv);
      h[s] = dA * h[s] + du * sB[t * 16 + s];
      y += h[s] * sC[t * 16 + s];
    }
    float zv = zp[(size_t)t * 128];
    float gate = zv / (1.0f + __expf(-zv));
    yp[(size_t)t * 128] = (y + uv * Dv) * gate;
  }
}

// ------------- Kernel 6: out = LN(y @ out_proj + xin) -------------
__global__ void __launch_bounds__(256) k_outln(
    const float* __restrict__ yb, const float* __restrict__ Wout,
    const float* __restrict__ xin, const float* __restrict__ gamma,
    const float* __restrict__ beta, float* __restrict__ out)
{
  __shared__ __align__(16) _Float16 sW[64 * 136];
  __shared__ float sRes[8 * 16 * 68];
  __shared__ float sG[64], sBt[64];
  int tid = threadIdx.x;
  for (int i = tid; i < 64 * 128; i += 256) {
    int n = i >> 7, k = i & 127;
    sW[n * 136 + k] = (_Float16)Wout[k * 64 + n];
  }
  if (tid < 64) { sG[tid] = gamma[tid]; sBt[tid] = beta[tid]; }
  __syncthreads();
  int wave = tid >> 5, lane = tid & 31;
  int row0 = blockIdx.x * 128 + wave * 16;
  float* res = sRes + wave * 16 * 68;
  v16h a0 = gldA(yb + (size_t)row0 * 128, 128, 0);
  v16h a1 = gldA(yb + (size_t)row0 * 128, 128, 32);
  v16h a2 = gldA(yb + (size_t)row0 * 128, 128, 64);
  v16h a3 = gldA(yb + (size_t)row0 * 128, 128, 96);
  int n0 = lane & 15, mb = (lane >> 4) * 8;
#pragma unroll
  for (int nt = 0; nt < 4; ++nt) {
    v8f acc = {};
    acc = wmma16(a0, ldsB(sW, 136, 0, nt), acc);
    acc = wmma16(a1, ldsB(sW, 136, 32, nt), acc);
    acc = wmma16(a2, ldsB(sW, 136, 64, nt), acc);
    acc = wmma16(a3, ldsB(sW, 136, 96, nt), acc);
#pragma unroll
    for (int r = 0; r < 8; ++r) {
      float rv = acc[r] + xin[(size_t)(row0 + mb + r) * 64 + nt * 16 + n0];
      res[(mb + r) * 68 + nt * 16 + n0] = rv;
    }
  }
  if (lane < 16) {  // one row per lane (same-wave LDS ops are in-order)
    float* rr = res + lane * 68;
    float mu = 0.f;
#pragma unroll
    for (int j = 0; j < 64; ++j) mu += rr[j];
    mu *= (1.0f / 64.0f);
    float var = 0.f;
#pragma unroll
    for (int j = 0; j < 64; ++j) { float dd = rr[j] - mu; var += dd * dd; }
    var *= (1.0f / 64.0f);
    float inv = rsqrtf(var + 0.001f);
    float* op = out + (size_t)(row0 + lane) * 64;
#pragma unroll
    for (int j = 0; j < 64; ++j) op[j] = sG[j] * (rr[j] - mu) * inv + sBt[j];
  }
}

extern "C" void kernel_launch(void* const* d_in, const int* in_sizes, int n_in,
                              void* d_out, int out_size, void* d_ws, size_t ws_size,
                              hipStream_t stream) {
  const float* x       = (const float*)d_in[0];
  const float* W_in    = (const float*)d_in[1];
  const float* b_in    = (const float*)d_in[2];
  const float* in_proj = (const float*)d_in[3];
  const float* conv_w  = (const float*)d_in[4];
  const float* conv_b  = (const float*)d_in[5];
  const float* x_proj  = (const float*)d_in[6];
  const float* dt_w    = (const float*)d_in[7];
  const float* dt_b    = (const float*)d_in[8];
  const float* A_log   = (const float*)d_in[9];
  const float* D_skip  = (const float*)d_in[10];
  const float* out_w   = (const float*)d_in[11];
  const float* gamma   = (const float*)d_in[12];
  const float* beta    = (const float*)d_in[13];
  float* out = (float*)d_out;

  float* ws = (float*)d_ws;
  const size_t M = (size_t)MTOT;
  float* xin = ws;                    // M*64
  float* xc  = xin + M * 64;          // M*128 (reused as y later)
  float* z   = xc + M * 128;          // M*128
  float* u   = z + M * 128;           // M*128
  float* dt  = u + M * 128;           // M*128
  float* Bm  = dt + M * 128;          // M*16
  float* Cm  = Bm + M * 16;           // M*16
  const size_t CS = (size_t)BB * NCHUNK * DINNER * DSTATE;
  float* P   = Cm + M * 16;           // CS
  float* he  = P + CS;                // CS
  float* hi  = he + CS;               // CS
  float* yb  = xc;                    // reuse: xc dead after k_conv_xproj

  k_inproj<<<MTOT / 128, 256, 0, stream>>>(x, W_in, b_in, in_proj, xin, xc, z);
  k_conv_xproj<<<MTOT / 64, 128, 0, stream>>>(xc, conv_w, conv_b, x_proj, dt_w, dt_b, u, dt, Bm, Cm);
  k_scan1<<<dim3(NCHUNK, BB), 128, 0, stream>>>(dt, u, Bm, A_log, P, he);
  k_combine<<<(BB * DINNER * DSTATE) / 256, 256, 0, stream>>>(P, he, hi);
  k_scan2<<<dim3(NCHUNK, BB), 128, 0, stream>>>(dt, u, Bm, Cm, A_log, D_skip, z, hi, yb);
  k_outln<<<MTOT / 128, 256, 0, stream>>>(yb, out_w, xin, gamma, beta, out);
}
